// RNN_10625749090471
// MI455X (gfx1250) — matrix operand order repaired and checked by reading, compiled, not run on hardware
//
#include <hip/hip_runtime.h>
#include <math.h>

// ---------------------------------------------------------------------------
// RNN: pre = X @ W_ih.T   (HBM-bound GEMM, split-precision f16 WMMA, f32 acc)
//      h_t = tanh(pre_t + b + W_hh h_{t-1})   (latency-bound scan, 1 wave32)
//      out = log_softmax(h_T @ fc_W.T + fc_b)
// ---------------------------------------------------------------------------

typedef __attribute__((ext_vector_type(16))) _Float16 v16h;
typedef __attribute__((ext_vector_type(8)))  float    v8f;

#define T_STEPS 100000
#define IN_DIM  256
#define HID     32
#define OUT_N   5

// ---------------------------------------------------------------------------
// Kernel A: pre[T][32] = X[T][256] @ W_ih[32][256]^T
// via V_WMMA_F32_16X16X32_F16 with Dekker split (x = f16hi + f16lo):
//   X@W ~= Xhi@Whi + Xhi@Wlo + Xlo@Whi   (lo*lo term ~2^-22, dropped)
// One wave per 16-row M tile; two N tiles (cols 0-15 / 16-31) share A frags.
// HBM roofline: 102.4 MB of X reads / 23.3 TB/s ~= 4.4 us; compute is free.
// ---------------------------------------------------------------------------
__global__ __launch_bounds__(256) void rnn_pre_gemm(const float* __restrict__ X,
                                                    const float* __restrict__ W_ih,
                                                    float* __restrict__ pre) {
  const int lane = threadIdx.x & 31;
  const int wave = threadIdx.x >> 5;
  const int tile = blockIdx.x * 8 + wave;
  const int t0   = tile * 16;
  if (t0 >= T_STEPS) return;

  const bool hi = lane >= 16;        // upper half-lanes carry the high K range
  const int  m  = t0 + (lane & 15);  // A row
  const int  n0 = lane & 15;         // B column (local)

  const float4* xr  = reinterpret_cast<const float4*>(X    + (size_t)m         * IN_DIM);
  const float4* w0r = reinterpret_cast<const float4*>(W_ih + (size_t)n0        * IN_DIM);
  const float4* w1r = reinterpret_cast<const float4*>(W_ih + (size_t)(n0 + 16) * IN_DIM);

  v8f acc0 = {};   // N = 0..15
  v8f acc1 = {};   // N = 16..31

  const int aO = hi ? 4 : 0;   // float4 offset of A's K base (hi?16:0 floats)
  const int bO = hi ? 4 : 0;   // float4 offset of B's K base (hi?16:0 floats)

#pragma unroll
  for (int kb = 0; kb < IN_DIM / 32; ++kb) {
    const int k4 = kb * 8;     // float4 index base of this 32-wide K block

    // A frag: K = base..base+15 of row m (base = hi?16:0)
    float4 fa0 = xr[k4 + aO + 0];
    float4 fa1 = xr[k4 + aO + 1];
    float4 fa2 = xr[k4 + aO + 2];
    float4 fa3 = xr[k4 + aO + 3];
    // B frags: 16 consecutive K of W_ih rows n0 / n0+16 (offset hi?16:0)
    float4 fb0 = w0r[k4 + bO + 0], fb1 = w0r[k4 + bO + 1];
    float4 fb2 = w0r[k4 + bO + 2], fb3 = w0r[k4 + bO + 3];
    float4 fc0 = w1r[k4 + bO + 0], fc1 = w1r[k4 + bO + 1];
    float4 fc2 = w1r[k4 + bO + 2], fc3 = w1r[k4 + bO + 3];

    float af[16] = {fa0.x, fa0.y, fa0.z, fa0.w,  fa1.x, fa1.y, fa1.z, fa1.w,
                    fa2.x, fa2.y, fa2.z, fa2.w,  fa3.x, fa3.y, fa3.z, fa3.w};
    float bf[16] = {fb0.x, fb0.y, fb0.z, fb0.w,  fb1.x, fb1.y, fb1.z, fb1.w,
                    fb2.x, fb2.y, fb2.z, fb2.w,  fb3.x, fb3.y, fb3.z, fb3.w};
    float cf[16] = {fc0.x, fc0.y, fc0.z, fc0.w,  fc1.x, fc1.y, fc1.z, fc1.w,
                    fc2.x, fc2.y, fc2.z, fc2.w,  fc3.x, fc3.y, fc3.z, fc3.w};

    v16h a_hi, a_lo, b0_hi, b0_lo, b1_hi, b1_lo;
#pragma unroll
    for (int i = 0; i < 16; ++i) {
      _Float16 h0 = (_Float16)af[i];  a_hi[i]  = h0;  a_lo[i]  = (_Float16)(af[i] - (float)h0);
      _Float16 h1 = (_Float16)bf[i];  b0_hi[i] = h1;  b0_lo[i] = (_Float16)(bf[i] - (float)h1);
      _Float16 h2 = (_Float16)cf[i];  b1_hi[i] = h2;  b1_lo[i] = (_Float16)(cf[i] - (float)h2);
    }

    acc0 = __builtin_amdgcn_wmma_f32_16x16x32_f16(false, a_hi, false, b0_hi, (short)0, acc0, false, false);
    acc1 = __builtin_amdgcn_wmma_f32_16x16x32_f16(false, a_hi, false, b1_hi, (short)0, acc1, false, false);
    acc0 = __builtin_amdgcn_wmma_f32_16x16x32_f16(false, a_hi, false, b0_lo, (short)0, acc0, false, false);
    acc1 = __builtin_amdgcn_wmma_f32_16x16x32_f16(false, a_hi, false, b1_lo, (short)0, acc1, false, false);
    acc0 = __builtin_amdgcn_wmma_f32_16x16x32_f16(false, a_lo, false, b0_hi, (short)0, acc0, false, false);
    acc1 = __builtin_amdgcn_wmma_f32_16x16x32_f16(false, a_lo, false, b1_hi, (short)0, acc1, false, false);
  }

  // D layout: VGPR r -> rows (r, r+8), lanes (0-15, 16-31), N = lane&15.
  const int rbase = t0 + (hi ? 8 : 0);
  const int col   = lane & 15;
#pragma unroll
  for (int r = 0; r < 8; ++r) {
    pre[(size_t)(rbase + r) * HID + col]      = acc0[r];
    pre[(size_t)(rbase + r) * HID + col + 16] = acc1[r];
  }
}

// ---------------------------------------------------------------------------
// tanh via V_EXP_F32 + fast divide; exact formula, arg of exp always <= 0.
// ---------------------------------------------------------------------------
__device__ __forceinline__ float tanh_fast(float x) {
  float ax = fabsf(x);
  float e  = __expf(-2.0f * ax);                  // v_exp_f32
  float r  = __fdividef(1.0f - e, 1.0f + e);      // v_rcp_f32 + mul
  return copysignf(r, x);
}

// Constant-lane broadcast that stays in the VALU domain (v_readlane_b32 ->
// SGPR -> v_fmac with scalar operand), avoiding the ds_bpermute LDS
// round-trip __shfl produces. Critical in a 100k-iteration serial loop.
__device__ __forceinline__ float bcast(float v, int lane) {
  return __int_as_float(__builtin_amdgcn_readlane(__float_as_int(v), lane));
}

// One recurrence step: h_new[i] = tanh(p + sum_j W_hh[i][j] * h[j])
__device__ __forceinline__ float rnn_step(const float* __restrict__ w,
                                          float p, float h) {
  float a0 = p, a1 = 0.0f, a2 = 0.0f, a3 = 0.0f;
#pragma unroll
  for (int j = 0; j < HID; j += 4) {
    a0 = fmaf(w[j + 0], bcast(h, j + 0), a0);
    a1 = fmaf(w[j + 1], bcast(h, j + 1), a1);
    a2 = fmaf(w[j + 2], bcast(h, j + 2), a2);
    a3 = fmaf(w[j + 3], bcast(h, j + 3), a3);
  }
  return tanh_fast((a0 + a1) + (a2 + a3));
}

// ---------------------------------------------------------------------------
// Kernel B: sequential scan, one wave32. Lane i owns W_hh row i (32 VGPRs)
// and hidden element h[i]. Per step: 32 readlane broadcasts feed 4
// independent FMA chains, then tanh. pre load software-pipelined one step
// ahead (12.8 MB stream, L2-resident). Tail: FC(5x32) + log_softmax via LDS.
// ---------------------------------------------------------------------------
__global__ __launch_bounds__(32) void rnn_scan(const float* __restrict__ pre,
                                               const float* __restrict__ W_hh,
                                               const float* __restrict__ b_ih,
                                               const float* __restrict__ b_hh,
                                               const float* __restrict__ fc_W,
                                               const float* __restrict__ fc_b,
                                               float* __restrict__ out) {
  const int lane = threadIdx.x;

  // lane i <- row i of W_hh, kept in registers
  float w[HID];
  const float4* wrow = reinterpret_cast<const float4*>(W_hh + (size_t)lane * HID);
#pragma unroll
  for (int j = 0; j < HID / 4; ++j) {
    float4 v = wrow[j];
    w[4 * j + 0] = v.x; w[4 * j + 1] = v.y;
    w[4 * j + 2] = v.z; w[4 * j + 3] = v.w;
  }

  const float bias = b_ih[lane] + b_hh[lane];
  const float* pp  = pre + lane;

  float h    = 0.0f;
  float pcur = pp[0] + bias;

  // Steady state: prefetch next step's pre unconditionally (t+1 < T here).
  for (int t = 0; t < T_STEPS - 1; ++t) {
    float pnext = pp[(size_t)(t + 1) * HID] + bias;
    h    = rnn_step(w, pcur, h);
    pcur = pnext;
  }
  h = rnn_step(w, pcur, h);   // final step, no prefetch

  // ---- FC + log_softmax tail ----
  __shared__ float hbuf[HID];
  __shared__ float lbuf[8];
  hbuf[lane] = h;
  __syncthreads();

  if (lane < OUT_N) {
    float acc = fc_b[lane];
#pragma unroll
    for (int i = 0; i < HID; ++i) acc = fmaf(fc_W[lane * HID + i], hbuf[i], acc);
    lbuf[lane] = acc;
  }
  __syncthreads();

  if (lane < OUT_N) {
    float mx = lbuf[0];
#pragma unroll
    for (int o = 1; o < OUT_N; ++o) mx = fmaxf(mx, lbuf[o]);
    float s = 0.0f;
#pragma unroll
    for (int o = 0; o < OUT_N; ++o) s += __expf(lbuf[o] - mx);
    out[lane] = lbuf[lane] - mx - __logf(s);
  }
}

// ---------------------------------------------------------------------------
extern "C" void kernel_launch(void* const* d_in, const int* in_sizes, int n_in,
                              void* d_out, int out_size, void* d_ws, size_t ws_size,
                              hipStream_t stream) {
  const float* X    = (const float*)d_in[0];
  const float* W_ih = (const float*)d_in[1];
  const float* W_hh = (const float*)d_in[2];
  const float* b_ih = (const float*)d_in[3];
  const float* b_hh = (const float*)d_in[4];
  const float* fc_W = (const float*)d_in[5];
  const float* fc_b = (const float*)d_in[6];
  float*       outp = (float*)d_out;
  float*       pre  = (float*)d_ws;   // needs T*32*4 = 12.8 MB of scratch

  const int tiles  = T_STEPS / 16;          // 6250 (exact)
  const int blocks = (tiles + 7) / 8;       // 8 waves / block
  rnn_pre_gemm<<<blocks, 256, 0, stream>>>(X, W_ih, pre);
  rnn_scan<<<1, 32, 0, stream>>>(pre, W_hh, b_ih, b_hh, fc_W, fc_b, outp);
}
